// SketchRNN_57526791962976
// MI455X (gfx1250) — compile-verified, matrix operand order checked by reading, not run on hardware
//
#include <hip/hip_runtime.h>
#include <hip/hip_bf16.h>

// ---------------------------------------------------------------------------
// SketchRNN-style VRNN for MI455X (gfx1250, wave32, WMMA).
// f16 storage for all GEMM operands, f32 WMMA accumulation.
// ---------------------------------------------------------------------------

typedef __attribute__((ext_vector_type(16))) _Float16 v16h;
typedef __attribute__((ext_vector_type(8)))  _Float16 v8h;
typedef __attribute__((ext_vector_type(8)))  float    v8f;

#define T_STEPS 192
#define BATCH   128
#define XIN_D   128
#define H_D     512
#define Z_D     128
#define OUT_D   64
#define G_D     1536          // 3*H
#define MROWS   (T_STEPS * BATCH)

// ---- math helpers ---------------------------------------------------------
__device__ __forceinline__ float sigmoid_f(float x) {
  return 1.0f / (1.0f + __expf(-x));
}
__device__ __forceinline__ float tanh_f(float x) {
  return 1.0f - 2.0f / (__expf(2.0f * x) + 1.0f);
}
__device__ __forceinline__ float softplus_f(float x) {
  return (x > 20.0f) ? x : __logf(1.0f + __expf(x));
}

// ---- WMMA fragment helpers ------------------------------------------------
// A-matrix 16x32 f16 layout (ISA 7.12.2): lane<16 holds row=lane, K={0..7,16..23};
// lane>=16 holds row=lane-16, K={8..15,24..31}. B (=W^T) loaded symmetrically
// from row-major W (N x K): lane indexes column n of B == row n of W.
__device__ __forceinline__ v16h ld_frag(const _Float16* base, int ld, int lane) {
  const _Float16* p = base + (size_t)(lane & 15) * ld + ((lane & 16) ? 8 : 0);
  v8h lo = *(const v8h*)(p);
  v8h hi = *(const v8h*)(p + 16);
  v16h r;
#pragma unroll
  for (int i = 0; i < 8; ++i) { r[i] = lo[i]; r[i + 8] = hi[i]; }
  return r;
}

__device__ __forceinline__ v8f wmma16(v16h a, v16h b, v8f c) {
  return __builtin_amdgcn_wmma_f32_16x16x32_f16(false, a, false, b, (short)0, c,
                                                false, false);
}

// ---- f32 -> f16 conversion ------------------------------------------------
__global__ void cvt_kernel(const float* __restrict__ s, _Float16* __restrict__ d,
                           int n) {
  int i = blockIdx.x * blockDim.x + threadIdx.x;
  if (i < n) d[i] = (_Float16)s[i];
}

// ---- generic tiled WMMA GEMM: C(M,N) = act(A(MxK) @ W(NxK)^T + bias) ------
// block = 256 threads (8 waves); block tile = 128(M) x 64(N); wave = 16x64.
// act: 0 none, 1 relu, 2 tanh, 3 softplus. Writes f32 (Cf) and/or f16 (Ch).
__global__ void gemm_wmma(const _Float16* __restrict__ A, int lda,
                          const _Float16* __restrict__ W, int ldw,
                          const float* __restrict__ bias,
                          float* __restrict__ Cf, _Float16* __restrict__ Ch,
                          int ldc, int M, int N, int K, int act) {
  const int lane = threadIdx.x & 31;
  const int wave = threadIdx.x >> 5;
  const int m0 = blockIdx.x * 128 + wave * 16;
  const int n0 = blockIdx.y * 64;
  if (m0 >= M) return;

  v8f acc0 = {}, acc1 = {}, acc2 = {}, acc3 = {};
  for (int k0 = 0; k0 < K; k0 += 32) {
    v16h a = ld_frag(A + (size_t)m0 * lda + k0, lda, lane);
    v16h b0 = ld_frag(W + (size_t)(n0 + 0) * ldw + k0, ldw, lane);
    acc0 = wmma16(a, b0, acc0);
    v16h b1 = ld_frag(W + (size_t)(n0 + 16) * ldw + k0, ldw, lane);
    acc1 = wmma16(a, b1, acc1);
    v16h b2 = ld_frag(W + (size_t)(n0 + 32) * ldw + k0, ldw, lane);
    acc2 = wmma16(a, b2, acc2);
    v16h b3 = ld_frag(W + (size_t)(n0 + 48) * ldw + k0, ldw, lane);
    acc3 = wmma16(a, b3, acc3);
  }

  // C/D layout: lanes 0-15 -> N=n, rows M=v; lanes 16-31 -> rows M=v+8.
  const int mb = m0 + ((lane & 16) ? 8 : 0);
  const int nc = lane & 15;
  v8f accs[4] = {acc0, acc1, acc2, acc3};
#pragma unroll
  for (int j = 0; j < 4; ++j) {
    const int n = n0 + j * 16 + nc;
    const float bv = bias ? bias[n] : 0.0f;
#pragma unroll
    for (int v = 0; v < 8; ++v) {
      float c = accs[j][v] + bv;
      if (act == 1)      c = fmaxf(c, 0.0f);
      else if (act == 2) c = tanh_f(c);
      else if (act == 3) c = softplus_f(c);
      const size_t idx = (size_t)(mb + v) * ldc + n;
      if (Cf) Cf[idx] = c;
      if (Ch) Ch[idx] = (_Float16)c;
    }
  }
}

// ---- persistent encoder GRU scan (one workgroup, 32 waves) ----------------
// h (128x512) f16 double-buffered in LDS; Whh streamed from L2; xp precomputed.
__global__ void gru_encoder(const _Float16* __restrict__ xp,   // (T*B, G)
                            const _Float16* __restrict__ Whh,  // (G, H)
                            const float* __restrict__ bhh,     // (G)
                            _Float16* __restrict__ hT,         // (B, H)
                            int reverse) {
  extern __shared__ _Float16 smem[];
  _Float16* hbuf0 = smem;
  _Float16* hbuf1 = smem + BATCH * H_D;
  for (int i = threadIdx.x; i < BATCH * H_D; i += blockDim.x)
    hbuf0[i] = (_Float16)0.0f;
  __syncthreads();

  const int lane = threadIdx.x & 31;
  const int wave = threadIdx.x >> 5;
  int cur = 0;

  for (int step = 0; step < T_STEPS; ++step) {
    const int t = reverse ? (T_STEPS - 1 - step) : step;
    const _Float16* h  = cur ? hbuf1 : hbuf0;
    _Float16*       hn = cur ? hbuf0 : hbuf1;
    const _Float16* xpt = xp + (size_t)t * BATCH * G_D;

    // 8 m-tiles x 32 n-tiles = 256 tile-triples; 8 per wave.
    for (int p = 0; p < 8; ++p) {
      const int pair = wave * 8 + p;
      const int m0 = (pair >> 5) * 16;
      const int n0 = (pair & 31) * 16;
      v8f aR = {}, aZ = {}, aN = {};
      for (int k0 = 0; k0 < H_D; k0 += 32) {
        v16h a = ld_frag(h + (size_t)m0 * H_D + k0, H_D, lane);
        aR = wmma16(a, ld_frag(Whh + (size_t)(n0) * H_D + k0, H_D, lane), aR);
        aZ = wmma16(a, ld_frag(Whh + (size_t)(H_D + n0) * H_D + k0, H_D, lane), aZ);
        aN = wmma16(a, ld_frag(Whh + (size_t)(2 * H_D + n0) * H_D + k0, H_D, lane), aN);
      }
      const int mb = m0 + ((lane & 16) ? 8 : 0);
      const int n  = n0 + (lane & 15);
      const float bR = bhh[n], bZ = bhh[H_D + n], bN = bhh[2 * H_D + n];
#pragma unroll
      for (int v = 0; v < 8; ++v) {
        const int m = mb + v;
        const float xr = (float)xpt[(size_t)m * G_D + n];
        const float xz = (float)xpt[(size_t)m * G_D + H_D + n];
        const float xn = (float)xpt[(size_t)m * G_D + 2 * H_D + n];
        const float r  = sigmoid_f(xr + aR[v] + bR);
        const float zg = sigmoid_f(xz + aZ[v] + bZ);
        const float nn = tanh_f(xn + r * (aN[v] + bN));
        const float ho = (float)h[(size_t)m * H_D + n];
        hn[(size_t)m * H_D + n] = (_Float16)((1.0f - zg) * nn + zg * ho);
      }
    }
    __syncthreads();
    cur ^= 1;
  }
  const _Float16* hf = cur ? hbuf1 : hbuf0;
  for (int i = threadIdx.x; i < BATCH * H_D; i += blockDim.x) hT[i] = hf[i];
}

// ---- persistent decoder GRU scan + output heads ---------------------------
// Note s == h_prev for t>0 (and s==0 at t=0), so r/z gates fuse s@Ws and
// h@Whh into one accumulator sharing the A fragment.
__global__ void gru_decoder(const float* __restrict__ zproj,   // (B,G) incl bih_d
                            const _Float16* __restrict__ Ws,   // (G,H), ld ldWs
                            int ldWs,
                            const _Float16* __restrict__ Whh,  // (G,H)
                            const float* __restrict__ bhh,
                            const _Float16* __restrict__ h0,   // (B,H) = dec_h0
                            const _Float16* __restrict__ Wdm,
                            const float* __restrict__ bdm,
                            const _Float16* __restrict__ Wds,
                            const float* __restrict__ bds,
                            const float* __restrict__ eps_y,   // (T,B,OUT)
                            float* __restrict__ preds,
                            float* __restrict__ means,
                            float* __restrict__ stds) {
  extern __shared__ _Float16 smem[];
  _Float16* hbuf0 = smem;
  _Float16* hbuf1 = smem + BATCH * H_D;
  for (int i = threadIdx.x; i < BATCH * H_D; i += blockDim.x) hbuf0[i] = h0[i];
  __syncthreads();

  const int lane = threadIdx.x & 31;
  const int wave = threadIdx.x >> 5;
  int cur = 0;

  for (int t = 0; t < T_STEPS; ++t) {
    const _Float16* h  = cur ? hbuf1 : hbuf0;
    _Float16*       hn = cur ? hbuf0 : hbuf1;

    // phase 1: h2 = GRUcell(h, zproj + s@Ws^T)
    for (int p = 0; p < 8; ++p) {
      const int pair = wave * 8 + p;
      const int m0 = (pair >> 5) * 16;
      const int n0 = (pair & 31) * 16;
      v8f aR = {}, aZ = {}, aXN = {}, aHN = {};
      for (int k0 = 0; k0 < H_D; k0 += 32) {
        v16h a = ld_frag(h + (size_t)m0 * H_D + k0, H_D, lane);
        aR  = wmma16(a, ld_frag(Whh + (size_t)(n0) * H_D + k0, H_D, lane), aR);
        aZ  = wmma16(a, ld_frag(Whh + (size_t)(H_D + n0) * H_D + k0, H_D, lane), aZ);
        aHN = wmma16(a, ld_frag(Whh + (size_t)(2 * H_D + n0) * H_D + k0, H_D, lane), aHN);
        if (t > 0) {  // s == h; at t==0, s==0 so the Ws terms vanish
          aR  = wmma16(a, ld_frag(Ws + (size_t)(n0) * ldWs + k0, ldWs, lane), aR);
          aZ  = wmma16(a, ld_frag(Ws + (size_t)(H_D + n0) * ldWs + k0, ldWs, lane), aZ);
          aXN = wmma16(a, ld_frag(Ws + (size_t)(2 * H_D + n0) * ldWs + k0, ldWs, lane), aXN);
        }
      }
      const int mb = m0 + ((lane & 16) ? 8 : 0);
      const int n  = n0 + (lane & 15);
      const float bR = bhh[n], bZ = bhh[H_D + n], bN = bhh[2 * H_D + n];
#pragma unroll
      for (int v = 0; v < 8; ++v) {
        const int m = mb + v;
        const float r  = sigmoid_f(zproj[(size_t)m * G_D + n] + aR[v] + bR);
        const float zg = sigmoid_f(zproj[(size_t)m * G_D + H_D + n] + aZ[v] + bZ);
        const float xn = zproj[(size_t)m * G_D + 2 * H_D + n] + aXN[v];
        const float nn = tanh_f(xn + r * (aHN[v] + bN));
        const float ho = (float)h[(size_t)m * H_D + n];
        hn[(size_t)m * H_D + n] = (_Float16)((1.0f - zg) * nn + zg * ho);
      }
    }
    __syncthreads();

    // phase 2: m/st heads from h2 (8x4 tiles x2 heads -> 1 tile-pair / wave)
    {
      const int m0 = (wave >> 2) * 16;
      const int n0 = (wave & 3) * 16;
      v8f aM = {}, aS = {};
      for (int k0 = 0; k0 < H_D; k0 += 32) {
        v16h a = ld_frag(hn + (size_t)m0 * H_D + k0, H_D, lane);
        aM = wmma16(a, ld_frag(Wdm + (size_t)n0 * H_D + k0, H_D, lane), aM);
        aS = wmma16(a, ld_frag(Wds + (size_t)n0 * H_D + k0, H_D, lane), aS);
      }
      const int mb = m0 + ((lane & 16) ? 8 : 0);
      const int n  = n0 + (lane & 15);
#pragma unroll
      for (int v = 0; v < 8; ++v) {
        const int m = mb + v;
        const float mm = softplus_f(aM[v] + bdm[n]);
        const float ss = softplus_f(aS[v] + bds[n]);
        const size_t oi = (size_t)t * BATCH * OUT_D + (size_t)m * OUT_D + n;
        means[oi] = mm;
        stds[oi]  = ss;
        preds[oi] = eps_y[oi] * ss + mm;
      }
    }
    __syncthreads();
    cur ^= 1;
  }
}

// ---- z = eps*std + mean; also build z_exp (B, 2Z) in f16 ------------------
__global__ void z_kernel(const float* __restrict__ eps_z,
                         const float* __restrict__ mean,
                         const float* __restrict__ stdv,
                         float* __restrict__ zout,
                         _Float16* __restrict__ zexp) {
  int i = blockIdx.x * blockDim.x + threadIdx.x;
  if (i >= 2 * BATCH * Z_D) return;
  const float z = eps_z[i] * stdv[i] + mean[i];
  zout[i] = z;
  const int s = i / (BATCH * Z_D);
  const int b = (i / Z_D) % BATCH;
  const int j = i % Z_D;
  zexp[(size_t)b * (2 * Z_D) + s * Z_D + j] = (_Float16)z;
}

// ---------------------------------------------------------------------------
extern "C" void kernel_launch(void* const* d_in, const int* in_sizes, int n_in,
                              void* d_out, int out_size, void* d_ws,
                              size_t ws_size, hipStream_t stream) {
  (void)in_sizes; (void)n_in; (void)out_size; (void)ws_size;

  const float* x      = (const float*)d_in[0];
  const float* eps_z  = (const float*)d_in[2];
  const float* eps_y  = (const float*)d_in[3];
  const float* W_phi1 = (const float*)d_in[4];
  const float* b_phi1 = (const float*)d_in[5];
  const float* W_phi2 = (const float*)d_in[6];
  const float* b_phi2 = (const float*)d_in[7];
  const float* Wih_ef = (const float*)d_in[8];
  const float* bih_ef = (const float*)d_in[9];
  const float* Whh_ef = (const float*)d_in[10];
  const float* bhh_ef = (const float*)d_in[11];
  const float* Wih_eb = (const float*)d_in[12];
  const float* bih_eb = (const float*)d_in[13];
  const float* Whh_eb = (const float*)d_in[14];
  const float* bhh_eb = (const float*)d_in[15];
  const float* W_mean = (const float*)d_in[16];
  const float* b_mean = (const float*)d_in[17];
  const float* W_std  = (const float*)d_in[18];
  const float* b_std  = (const float*)d_in[19];
  const float* W_fh   = (const float*)d_in[20];
  const float* b_fh   = (const float*)d_in[21];
  const float* Wih_d  = (const float*)d_in[22];
  const float* bih_d  = (const float*)d_in[23];
  const float* Whh_d  = (const float*)d_in[24];
  const float* bhh_d  = (const float*)d_in[25];
  const float* W_dm   = (const float*)d_in[26];
  const float* b_dm   = (const float*)d_in[27];
  const float* W_ds   = (const float*)d_in[28];
  const float* b_ds   = (const float*)d_in[29];

  // output layout (f32, flattened in return order)
  float* out    = (float*)d_out;
  float* o_mean = out;                                 // (2,128,128)
  float* o_std  = out + 32768;                         // (2,128,128)
  float* o_z    = out + 65536;                         // (2,128,128)
  float* o_pred = out + 98304;                         // (192,128,64)
  float* o_m    = out + 98304 + 1572864;
  float* o_s    = out + 98304 + 2 * 1572864;

  // workspace carve-out
  char* ws = (char*)d_ws;
  size_t off = 0;
  auto alloc = [&](size_t bytes) -> char* {
    off = (off + 255) & ~(size_t)255;
    char* p = ws + off;
    off += bytes;
    return p;
  };
  _Float16* x16    = (_Float16*)alloc((size_t)MROWS * XIN_D * 2);
  _Float16* xp16   = (_Float16*)alloc((size_t)MROWS * G_D * 2);  // also t1 alias
  _Float16* phi16  = (_Float16*)alloc((size_t)MROWS * H_D * 2);
  _Float16* wphi1  = (_Float16*)alloc((size_t)H_D * XIN_D * 2);
  _Float16* wphi2  = (_Float16*)alloc((size_t)H_D * H_D * 2);
  _Float16* wihef  = (_Float16*)alloc((size_t)G_D * H_D * 2);
  _Float16* whhef  = (_Float16*)alloc((size_t)G_D * H_D * 2);
  _Float16* wiheb  = (_Float16*)alloc((size_t)G_D * H_D * 2);
  _Float16* whheb  = (_Float16*)alloc((size_t)G_D * H_D * 2);
  _Float16* wmean  = (_Float16*)alloc((size_t)Z_D * H_D * 2);
  _Float16* wstd   = (_Float16*)alloc((size_t)Z_D * H_D * 2);
  _Float16* wfh    = (_Float16*)alloc((size_t)H_D * 2 * Z_D * 2);
  _Float16* wihd   = (_Float16*)alloc((size_t)G_D * (2 * Z_D + H_D) * 2);
  _Float16* whhd   = (_Float16*)alloc((size_t)G_D * H_D * 2);
  _Float16* wdm    = (_Float16*)alloc((size_t)OUT_D * H_D * 2);
  _Float16* wds    = (_Float16*)alloc((size_t)OUT_D * H_D * 2);
  _Float16* ench16 = (_Float16*)alloc((size_t)2 * BATCH * H_D * 2);
  _Float16* zexp16 = (_Float16*)alloc((size_t)BATCH * 2 * Z_D * 2);
  _Float16* dech0  = (_Float16*)alloc((size_t)BATCH * H_D * 2);
  float*    zproj  = (float*)alloc((size_t)BATCH * G_D * 4);

  auto cvt = [&](const float* s, _Float16* d, int n) {
    cvt_kernel<<<(n + 255) / 256, 256, 0, stream>>>(s, d, n);
  };
  auto gemm = [&](const _Float16* A, int lda, const _Float16* W, int ldw,
                  const float* bias, float* Cf, _Float16* Ch, int ldc, int M,
                  int N, int K, int act) {
    dim3 grid((M + 127) / 128, N / 64);
    gemm_wmma<<<grid, 256, 0, stream>>>(A, lda, W, ldw, bias, Cf, Ch, ldc, M, N,
                                        K, act);
  };

  // f32 -> f16 conversions
  cvt(x, x16, MROWS * XIN_D);
  cvt(W_phi1, wphi1, H_D * XIN_D);
  cvt(W_phi2, wphi2, H_D * H_D);
  cvt(Wih_ef, wihef, G_D * H_D);
  cvt(Whh_ef, whhef, G_D * H_D);
  cvt(Wih_eb, wiheb, G_D * H_D);
  cvt(Whh_eb, whheb, G_D * H_D);
  cvt(W_mean, wmean, Z_D * H_D);
  cvt(W_std, wstd, Z_D * H_D);
  cvt(W_fh, wfh, H_D * 2 * Z_D);
  cvt(Wih_d, wihd, G_D * (2 * Z_D + H_D));
  cvt(Whh_d, whhd, G_D * H_D);
  cvt(W_dm, wdm, OUT_D * H_D);
  cvt(W_ds, wds, OUT_D * H_D);

  const size_t GRU_LDS = (size_t)2 * BATCH * H_D * sizeof(_Float16);  // 256 KB
  (void)hipFuncSetAttribute((const void*)gru_encoder,
                            hipFuncAttributeMaxDynamicSharedMemorySize,
                            (int)GRU_LDS);
  (void)hipFuncSetAttribute((const void*)gru_decoder,
                            hipFuncAttributeMaxDynamicSharedMemorySize,
                            (int)GRU_LDS);

  // phi MLP: t1 (aliased into xp16) then phi16
  gemm(x16, XIN_D, wphi1, XIN_D, b_phi1, nullptr, xp16, H_D, MROWS, H_D, XIN_D, 1);
  gemm(xp16, H_D, wphi2, H_D, b_phi2, nullptr, phi16, H_D, MROWS, H_D, H_D, 1);

  // forward encoder: x-projection then scan
  gemm(phi16, H_D, wihef, H_D, bih_ef, nullptr, xp16, G_D, MROWS, G_D, H_D, 0);
  gru_encoder<<<1, 1024, GRU_LDS, stream>>>(xp16, whhef, bhh_ef, ench16, 0);

  // backward encoder: reuse xp16 buffer, iterate reversed
  gemm(phi16, H_D, wiheb, H_D, bih_eb, nullptr, xp16, G_D, MROWS, G_D, H_D, 0);
  gru_encoder<<<1, 1024, GRU_LDS, stream>>>(xp16, whheb, bhh_eb,
                                            ench16 + (size_t)BATCH * H_D, 1);

  // latent heads: mean (identity), std (softplus) -> straight into d_out
  gemm(ench16, H_D, wmean, H_D, b_mean, o_mean, nullptr, Z_D, 2 * BATCH, Z_D, H_D, 0);
  gemm(ench16, H_D, wstd, H_D, b_std, o_std, nullptr, Z_D, 2 * BATCH, Z_D, H_D, 3);

  // z = eps*std + mean, plus z_exp f16
  z_kernel<<<(2 * BATCH * Z_D + 255) / 256, 256, 0, stream>>>(eps_z, o_mean,
                                                              o_std, o_z, zexp16);

  // dec_h0 = tanh(z_exp @ W_fh^T + b_fh); z_proj = z_exp @ Wz^T + bih_d
  gemm(zexp16, 2 * Z_D, wfh, 2 * Z_D, b_fh, nullptr, dech0, H_D, BATCH, H_D,
       2 * Z_D, 2);
  gemm(zexp16, 2 * Z_D, wihd, 2 * Z_D + H_D, bih_d, zproj, nullptr, G_D, BATCH,
       G_D, 2 * Z_D, 0);

  // decoder scan (Ws = Wih_d columns [2Z, 2Z+H), row stride 768)
  gru_decoder<<<1, 1024, GRU_LDS, stream>>>(zproj, wihd + 2 * Z_D, 2 * Z_D + H_D,
                                            whhd, bhh_d, dech0, wdm, b_dm, wds,
                                            b_ds, eps_y, o_pred, o_m, o_s);
}
